// YomiHandChoiceModel_38190849196520
// MI455X (gfx1250) — matrix operand order repaired and checked by reading, not compiled, not run on hardware
//
#include <hip/hip_runtime.h>
#include <math.h>

typedef __attribute__((ext_vector_type(16))) __bf16 v16bf;
typedef __attribute__((ext_vector_type(8)))  __bf16 v8bf;
typedef __attribute__((ext_vector_type(8)))  float  v8f;

// Problem constants (match reference)
constexpr int NB   = 1024;          // graphs
constexpr int NPGc = 12;            // nodes per graph
constexpr int NN   = NB * NPGc;     // 12288 nodes per type
constexpr int EPGc = 72;            // edges per graph per type
constexpr int EE   = NB * EPGc;     // 73728 edges per type
constexpr int HIDc = 256;
constexpr int NHds = 4;
constexpr int LAY  = 3;
constexpr int FINc = 128;

// ---------------- WMMA fragment loaders (ISA 7.12.2 layouts) ----------------
// A: 16x32 bf16, row-major bf16 source. lane l: m = l&15, half = l>>4.
//   e in [0,8):  K = k0 + 8*half + e          -> one 16B load
//   e in [8,16): K = k0 + 16 + 8*half + (e-8) -> one 16B load
__device__ __forceinline__ v16bf load_afrag_bf(const __bf16* __restrict__ A, int lda,
                                               int row, int k0, int half) {
  const __bf16* p = A + (size_t)row * lda + k0 + 8 * half;
  v8bf lo = *(const v8bf*)p;
  v8bf hi = *(const v8bf*)(p + 16);
  return __builtin_shufflevector(lo, hi, 0, 1, 2, 3, 4, 5, 6, 7,
                                 8, 9, 10, 11, 12, 13, 14, 15);
}

// ---------------- elementwise fp32 -> bf16 (row-major shadow) ----------------
__global__ void cvt_row_kernel(const float* __restrict__ in, __bf16* __restrict__ out, int n) {
  int i = blockIdx.x * blockDim.x + threadIdx.x;
  if (i < n) out[i] = (__bf16)in[i];
}

// ---------------- weight fp32 -> bf16 fragment-linear pre-swizzle ----------------
// Tile t = tn*(K/32) + ks holds the 32x16 B-tile (k0=ks*32, n0=tn*16) as
// 32 lanes x 16 contiguous bf16: lane l -> col n0+(l&15), k = k0 + 8*(l>>4) + (e&7) + 16*(e>>3).
__global__ void cvt_tiled_kernel(const float* __restrict__ in, __bf16* __restrict__ out,
                                 int K, int N, int nmat) {
  int idx = blockIdx.x * blockDim.x + threadIdx.x;
  if (idx >= nmat * K * N) return;
  int mat = idx / (K * N);
  int r   = idx % (K * N);
  int tile   = r >> 9;
  int within = r & 511;
  int lane = within >> 4;
  int e    = within & 15;
  int ksteps = K >> 5;
  int ks = tile % ksteps;
  int tn = tile / ksteps;
  int n = tn * 16 + (lane & 15);
  int k = ks * 32 + 8 * (lane >> 4) + (e & 7) + 16 * (e >> 3);
  out[idx] = (__bf16)in[(size_t)mat * K * N + (size_t)k * N + n];
}

// ---------------- Generic GEMM: out[M,N] = A[M,K] @ W[K,N] + bias ----------------
// A is a row-major bf16 shadow; Wt is fragment-linear tiled bf16.
// One wave computes a 64x16 strip (4 M-tiles sharing each B fragment).
// Writes fp32 out and (optionally) a bf16 row-major shadow outb.
// act: 0 = none, 1 = leaky_relu(slope)
__global__ void gemm_wmma_kernel(const __bf16* __restrict__ A, const __bf16* __restrict__ Wt,
                                 const float* __restrict__ bias, float* __restrict__ out,
                                 __bf16* __restrict__ outb,
                                 int M, int Nout, int K, int act, float slope) {
  int wave = blockIdx.x * 8 + (threadIdx.x >> 5);
  int lane = threadIdx.x & 31;
  int ntiles = Nout >> 4;
  int sm = wave / ntiles, tn = wave % ntiles;
  if (sm * 64 >= M) return;
  int half = lane >> 4, li = lane & 15;
  int ncol = tn * 16 + li;
  int m0 = sm * 64;
  int ksteps = K >> 5;

  const __bf16* pb = Wt + ((size_t)tn * ksteps) * 512 + lane * 16;

  v8f acc[4] = {{}, {}, {}, {}};
  for (int ks = 0; ks < ksteps; ++ks, pb += 512) {
    if (ks + 1 < ksteps)  // next 1KB weight tile (global_prefetch_b8)
      __builtin_prefetch(pb + 512, 0, 0);
    v16bf b = *(const v16bf*)pb;   // 32B contiguous -> 2x global_load_b128
    int k0 = ks * 32;
#pragma unroll
    for (int mi = 0; mi < 4; ++mi) {
      v16bf a = load_afrag_bf(A, K, m0 + mi * 16 + li, k0, half);
      acc[mi] = __builtin_amdgcn_wmma_f32_16x16x32_bf16(false, a, false, b, (short)0,
                                                        acc[mi], false, false);
    }
  }
  float bv = bias ? bias[ncol] : 0.0f;
#pragma unroll
  for (int mi = 0; mi < 4; ++mi)
#pragma unroll
    for (int r = 0; r < 8; ++r) {
      float v = acc[mi][r] + bv;
      if (act == 1) v = (v > 0.0f) ? v : slope * v;
      size_t o = (size_t)(m0 + mi * 16 + r + 8 * half) * Nout + ncol;
      out[o] = v;
      if (outb) outb[o] = (__bf16)v;
    }
}

// ---------------- Fused CGConv ----------------
// z_e = [h_dst[dst[e]], h_src[src[e]]] (512, bf16 shadows);
// msg = sigmoid(z@Wf+bf)*softplus(z@Ws+bs), scattered into fp32 nbuf[dst[e], :].
// One wave = 64 edges x 16 cols (4 edge-tiles, Wf+Ws).
__global__ void cgconv_wmma_kernel(const __bf16* __restrict__ hsrcb,
                                   const __bf16* __restrict__ hdstb,
                                   const int* __restrict__ src, const int* __restrict__ dst,
                                   const __bf16* __restrict__ Wf, const float* __restrict__ bf_,
                                   const __bf16* __restrict__ Ws, const float* __restrict__ bs_,
                                   float* __restrict__ nbuf) {
  int wave = blockIdx.x * 8 + (threadIdx.x >> 5);
  int lane = threadIdx.x & 31;
  int sm = wave >> 4;        // 16 col tiles (256/16)
  int tn = wave & 15;
  if (sm * 64 >= EE) return;
  int half = lane >> 4, li = lane & 15;
  int ncol = tn * 16 + li;
  int m0 = sm * 64;
  constexpr int KST = (2 * HIDc) >> 5;   // 16 k-steps

  const __bf16* pd[4];
  const __bf16* ps[4];
#pragma unroll
  for (int mi = 0; mi < 4; ++mi) {
    int erow = m0 + mi * 16 + li;
    pd[mi] = hdstb + (size_t)dst[erow] * HIDc;
    ps[mi] = hsrcb + (size_t)src[erow] * HIDc;
  }

  const __bf16* pbF = Wf + ((size_t)tn * KST) * 512 + lane * 16;
  const __bf16* pbS = Ws + ((size_t)tn * KST) * 512 + lane * 16;

  v8f accF[4] = {{}, {}, {}, {}};
  v8f accS[4] = {{}, {}, {}, {}};
  for (int ks = 0; ks < KST; ++ks, pbF += 512, pbS += 512) {
    if (ks + 1 < KST) {
      __builtin_prefetch(pbF + 512, 0, 0);
      __builtin_prefetch(pbS + 512, 0, 0);
    }
    v16bf bF = *(const v16bf*)pbF;
    v16bf bS = *(const v16bf*)pbS;
    int kb = ks * 32 + 8 * half;       // multiple of 8; runs never split the 256 boundary
    int k2 = kb + 16;
#pragma unroll
    for (int mi = 0; mi < 4; ++mi) {
      const __bf16* q1 = (kb < HIDc) ? pd[mi] + kb : ps[mi] + (kb - HIDc);
      const __bf16* q2 = (k2 < HIDc) ? pd[mi] + k2 : ps[mi] + (k2 - HIDc);
      v8bf lo = *(const v8bf*)q1;
      v8bf hi = *(const v8bf*)q2;
      v16bf a = __builtin_shufflevector(lo, hi, 0, 1, 2, 3, 4, 5, 6, 7,
                                        8, 9, 10, 11, 12, 13, 14, 15);
      accF[mi] = __builtin_amdgcn_wmma_f32_16x16x32_bf16(false, a, false, bF, (short)0,
                                                         accF[mi], false, false);
      accS[mi] = __builtin_amdgcn_wmma_f32_16x16x32_bf16(false, a, false, bS, (short)0,
                                                         accS[mi], false, false);
    }
  }
  float bfv = bf_[ncol], bsv = bs_[ncol];
#pragma unroll
  for (int mi = 0; mi < 4; ++mi)
#pragma unroll
    for (int r = 0; r < 8; ++r) {
      int e = m0 + mi * 16 + r + 8 * half;
      float gF = accF[mi][r] + bfv;
      float gS = accS[mi][r] + bsv;
      float sg = 1.0f / (1.0f + __expf(-gF));
      float sp = (gS > 20.0f) ? gS : log1pf(__expf(gS));
      atomicAdd(&nbuf[(size_t)dst[e] * HIDc + tn * 16 + (lane & 15)], sg * sp);
    }
}

// ---------------- GATv2 attention pieces (fp32) ----------------
__global__ void gat_alpha_kernel(const float* __restrict__ xl, const float* __restrict__ xr,
                                 const int* __restrict__ src, const int* __restrict__ dst,
                                 const float* __restrict__ att, float* __restrict__ alpha) {
  int wave = blockIdx.x * 8 + (threadIdx.x >> 5);
  int lane = threadIdx.x & 31;
  if (wave >= EE * NHds) return;
  int e = wave >> 2, h = wave & 3;
  const float* pl = xl + (size_t)src[e] * (NHds * HIDc) + h * HIDc;
  const float* pr = xr + (size_t)dst[e] * (NHds * HIDc) + h * HIDc;
  const float* pa = att + h * HIDc;
  float sum = 0.0f;
  for (int c = lane; c < HIDc; c += 32) {
    float t = pl[c] + pr[c];
    t = (t > 0.0f) ? t : 0.2f * t;
    sum += t * pa[c];
  }
#pragma unroll
  for (int off = 16; off; off >>= 1) sum += __shfl_xor(sum, off, 32);
  if (lane == 0) alpha[wave] = sum;
}

__global__ void gat_maxdenom_kernel(const float* __restrict__ alpha, const int* __restrict__ dst,
                                    float* __restrict__ mbuf, float* __restrict__ dbuf) {
  int idx = blockIdx.x * blockDim.x + threadIdx.x;
  if (idx >= NN * NHds) return;
  int i = idx >> 2, h = idx & 3;
  int e0 = (i / NPGc) * EPGc;
  float m = -3.0e38f;
  for (int e = e0; e < e0 + EPGc; ++e)
    if (dst[e] == i) { float a = alpha[e * NHds + h]; m = (a > m) ? a : m; }
  float ds = 0.0f;
  for (int e = e0; e < e0 + EPGc; ++e)
    if (dst[e] == i) ds += __expf(alpha[e * NHds + h] - m);
  mbuf[idx] = m;
  dbuf[idx] = ds;
}

__global__ void gat_weight_kernel(float* __restrict__ alpha, const int* __restrict__ dst,
                                  const float* __restrict__ mbuf, const float* __restrict__ dbuf) {
  int idx = blockIdx.x * blockDim.x + threadIdx.x;
  if (idx >= EE * NHds) return;
  int e = idx >> 2, h = idx & 3;
  int di = dst[e] * NHds + h;
  alpha[idx] = __expf(alpha[idx] - mbuf[di]) / (dbuf[di] + 1e-16f) * 0.25f;
}

__global__ void gat_scatter_kernel(const float* __restrict__ w, const float* __restrict__ xl,
                                   const int* __restrict__ src, const int* __restrict__ dst,
                                   float* __restrict__ nbuf) {
  int idx = blockIdx.x * blockDim.x + threadIdx.x;
  if (idx >= EE * HIDc) return;
  int e = idx >> 8, c = idx & 255;
  const float* pl = xl + (size_t)src[e] * (NHds * HIDc) + c;
  float acc = 0.0f;
#pragma unroll
  for (int h = 0; h < NHds; ++h) acc += w[e * NHds + h] * pl[h * HIDc];
  atomicAdd(&nbuf[(size_t)dst[e] * HIDc + c], acc);
}

__global__ void init_new_kernel(float* __restrict__ nbuf, const float* __restrict__ hdst,
                                const float* __restrict__ gbias) {
  int idx = blockIdx.x * blockDim.x + threadIdx.x;
  if (idx >= NN * HIDc) return;
  nbuf[idx] = hdst[idx] + gbias[idx & 255];
}

__global__ void final_kernel(const float* __restrict__ f, const float* __restrict__ W2,
                             const float* __restrict__ b2, float* __restrict__ out) {
  int i = blockIdx.x * blockDim.x + threadIdx.x;
  if (i >= NN) return;
  const float* pf = f + (size_t)i * FINc;
  float s = b2[0];
  for (int c = 0; c < FINc; ++c) s += pf[c] * W2[c];
  out[i] = s;
}

// ---------------- host orchestration ----------------
static void launch_gemm(const __bf16* A, const __bf16* Wt, const float* bias, float* out,
                        __bf16* outb, int M, int Nout, int K, int act, float slope,
                        hipStream_t s) {
  int strips = (M / 64) * (Nout / 16);
  int blocks = (strips + 7) / 8;
  gemm_wmma_kernel<<<blocks, 256, 0, s>>>(A, Wt, bias, out, outb, M, Nout, K, act, slope);
}

static void launch_cvt_tiled(const float* in, __bf16* out, int K, int N, int nmat,
                             hipStream_t s) {
  int n = nmat * K * N;
  cvt_tiled_kernel<<<(n + 255) / 256, 256, 0, s>>>(in, out, K, N, nmat);
}

static void launch_cvt_row(const float* in, __bf16* out, int n, hipStream_t s) {
  cvt_row_kernel<<<(n + 255) / 256, 256, 0, s>>>(in, out, n);
}

extern "C" void kernel_launch(void* const* d_in, const int* in_sizes, int n_in,
                              void* d_out, int out_size, void* d_ws, size_t ws_size,
                              hipStream_t stream) {
  const float* x_my    = (const float*)d_in[0];
  const float* x_opp   = (const float*)d_in[1];
  const int*   e_beats = (const int*)d_in[2];
  const int*   e_loses = (const int*)d_in[3];
  const int*   e_rbeat = (const int*)d_in[4];
  const int*   e_rlose = (const int*)d_in[5];
  // d_in[6] = batch_my (unused; dense layout == node order)
  const float* enc_W   = (const float*)d_in[7];
  const float* enc_b   = (const float*)d_in[8];
  const float* gat_Wl  = (const float*)d_in[9];
  const float* gat_bl  = (const float*)d_in[10];
  const float* gat_Wr  = (const float*)d_in[11];
  const float* gat_br  = (const float*)d_in[12];
  const float* gat_att = (const float*)d_in[13];
  const float* gat_bias= (const float*)d_in[14];
  const float* cg_Wf   = (const float*)d_in[15];
  const float* cg_bf   = (const float*)d_in[16];
  const float* cg_Ws   = (const float*)d_in[17];
  const float* cg_bs   = (const float*)d_in[18];
  const float* node_W  = (const float*)d_in[19];
  const float* node_b  = (const float*)d_in[20];
  const float* fin_W1  = (const float*)d_in[21];
  const float* fin_b1  = (const float*)d_in[22];
  const float* fin_W2  = (const float*)d_in[23];
  const float* fin_b2  = (const float*)d_in[24];

  // -------- workspace carve: fp32 buffers, then bf16 shadows --------
  float* w = (float*)d_ws;
  float* h_my   = w; w += (size_t)NN * HIDc;
  float* h_opp  = w; w += (size_t)NN * HIDc;
  float* new_my = w; w += (size_t)NN * HIDc;
  float* new_opp= w; w += (size_t)NN * HIDc;
  float* xl     = w; w += (size_t)NN * NHds * HIDc;
  float* xr     = w; w += (size_t)NN * NHds * HIDc;
  float* alpha  = w; w += (size_t)EE * NHds;
  float* mbuf   = w; w += (size_t)NN * NHds;
  float* dbuf   = w; w += (size_t)NN * NHds;
  float* fbuf   = new_my;   // reuse after layer loop

  constexpr int SZ_ENC  = 32 * HIDc;                       // 8192
  constexpr int SZ_GATW = LAY * 2 * HIDc * NHds * HIDc;    // 1572864 (each of Wl, Wr)
  constexpr int SZ_CGW  = LAY * 2 * 2 * HIDc * HIDc;       // 786432  (each of Wf, Ws)
  constexpr int SZ_NODE = LAY * HIDc * HIDc;               // 196608
  constexpr int SZ_FIN1 = HIDc * FINc;                     // 32768

  __bf16* bw = (__bf16*)w;
  // tiled (fragment-linear) weight shadows
  __bf16* enc_Wb  = bw; bw += SZ_ENC;
  __bf16* gat_Wlb = bw; bw += SZ_GATW;
  __bf16* gat_Wrb = bw; bw += SZ_GATW;
  __bf16* cg_Wfb  = bw; bw += SZ_CGW;
  __bf16* cg_Wsb  = bw; bw += SZ_CGW;
  __bf16* node_Wb = bw; bw += SZ_NODE;
  __bf16* fin_W1b = bw; bw += SZ_FIN1;
  // row-major activation shadows
  __bf16* x_myb   = bw; bw += (size_t)NN * 32;
  __bf16* x_oppb  = bw; bw += (size_t)NN * 32;
  __bf16* h_myb   = bw; bw += (size_t)NN * HIDc;
  __bf16* h_oppb  = bw; bw += (size_t)NN * HIDc;
  __bf16* new_myb = bw; bw += (size_t)NN * HIDc;
  __bf16* new_oppb= bw; bw += (size_t)NN * HIDc;

  // weight pre-swizzle + input shadows
  launch_cvt_tiled(enc_W,  enc_Wb,  32,   HIDc,        1,       stream);
  launch_cvt_tiled(gat_Wl, gat_Wlb, HIDc, NHds * HIDc, LAY * 2, stream);
  launch_cvt_tiled(gat_Wr, gat_Wrb, HIDc, NHds * HIDc, LAY * 2, stream);
  launch_cvt_tiled(cg_Wf,  cg_Wfb,  2 * HIDc, HIDc,    LAY * 2, stream);
  launch_cvt_tiled(cg_Ws,  cg_Wsb,  2 * HIDc, HIDc,    LAY * 2, stream);
  launch_cvt_tiled(node_W, node_Wb, HIDc, HIDc,        LAY,     stream);
  launch_cvt_tiled(fin_W1, fin_W1b, HIDc, FINc,        1,       stream);
  launch_cvt_row(x_my,  x_myb,  NN * 32, stream);
  launch_cvt_row(x_opp, x_oppb, NN * 32, stream);

  // input encoder (writes fp32 h + bf16 shadow)
  launch_gemm(x_myb,  enc_Wb, enc_b, h_my,  h_myb,  NN, HIDc, 32, 0, 0.0f, stream);
  launch_gemm(x_oppb, enc_Wb, enc_b, h_opp, h_oppb, NN, HIDc, 32, 0, 0.0f, stream);

  for (int l = 0; l < LAY; ++l) {
    for (int dir = 0; dir < 2; ++dir) {
      const float*  hdst  = (dir == 0) ? h_opp  : h_my;
      const __bf16* hsrcb = (dir == 0) ? h_myb  : h_oppb;
      const __bf16* hdstb = (dir == 0) ? h_oppb : h_myb;
      float*        nbuf  = (dir == 0) ? new_opp : new_my;
      const int*    ge    = (dir == 0) ? e_beats : e_rlose;  // GAT edges
      const int*    ce    = (dir == 0) ? e_loses : e_rbeat;  // CGConv edges
      size_t pi = (size_t)(l * 2 + dir);

      init_new_kernel<<<(NN * HIDc) / 256, 256, 0, stream>>>(nbuf, hdst, gat_bias + pi * HIDc);

      // GATv2: xl = hsrc@Wl+bl, xr = hdst@Wr+br   [NN, 1024]
      launch_gemm(hsrcb, gat_Wlb + pi * HIDc * NHds * HIDc, gat_bl + pi * NHds * HIDc,
                  xl, nullptr, NN, NHds * HIDc, HIDc, 0, 0.0f, stream);
      launch_gemm(hdstb, gat_Wrb + pi * HIDc * NHds * HIDc, gat_br + pi * NHds * HIDc,
                  xr, nullptr, NN, NHds * HIDc, HIDc, 0, 0.0f, stream);
      gat_alpha_kernel<<<(EE * NHds) / 8, 256, 0, stream>>>(xl, xr, ge, ge + EE,
                                                            gat_att + pi * NHds * HIDc, alpha);
      gat_maxdenom_kernel<<<(NN * NHds) / 256, 256, 0, stream>>>(alpha, ge + EE, mbuf, dbuf);
      gat_weight_kernel<<<(EE * NHds) / 256, 256, 0, stream>>>(alpha, ge + EE, mbuf, dbuf);
      gat_scatter_kernel<<<(EE * HIDc) / 256, 256, 0, stream>>>(alpha, xl, ge, ge + EE, nbuf);

      // CGConv (fused bf16 gather + dual GEMM + gate + scatter)
      cgconv_wmma_kernel<<<((EE / 64) * 16) / 8, 256, 0, stream>>>(
          hsrcb, hdstb, ce, ce + EE,
          cg_Wfb + pi * 2 * HIDc * HIDc, cg_bf + pi * HIDc,
          cg_Wsb + pi * 2 * HIDc * HIDc, cg_bs + pi * HIDc, nbuf);
    }
    // shadow the scattered accumulators, then nodewise linear
    launch_cvt_row(new_my,  new_myb,  NN * HIDc, stream);
    launch_cvt_row(new_opp, new_oppb, NN * HIDc, stream);
    launch_gemm(new_myb,  node_Wb + (size_t)l * HIDc * HIDc, node_b + (size_t)l * HIDc,
                h_my,  h_myb,  NN, HIDc, HIDc, 0, 0.0f, stream);
    launch_gemm(new_oppb, node_Wb + (size_t)l * HIDc * HIDc, node_b + (size_t)l * HIDc,
                h_opp, h_oppb, NN, HIDc, HIDc, 0, 0.0f, stream);
  }

  // final encoder: leaky_relu slope = 128.0 (faithful to reference bug)
  launch_gemm(h_myb, fin_W1b, fin_b1, fbuf, nullptr, NN, FINc, HIDc, 1, 128.0f, stream);
  final_kernel<<<NN / 256, 256, 0, stream>>>(fbuf, fin_W2, fin_b2, (float*)d_out);
}
